// MSGAD_214748365383
// MI455X (gfx1250) — matrix operand order repaired and checked by reading, not compiled, not run on hardware
//
#include <hip/hip_runtime.h>

typedef __attribute__((ext_vector_type(2)))  float     v2f;
typedef __attribute__((ext_vector_type(8)))  float     v8f;
typedef __attribute__((ext_vector_type(8)))  _Float16  v8h;
typedef __attribute__((ext_vector_type(16))) _Float16  v16h;

// ---------------------------------------------------------------------------
// Tiny transpose: in is R x C row-major, out is C x R row-major (out = in^T)
// ---------------------------------------------------------------------------
__global__ void MSGAD_transpose_kernel(const float* __restrict__ in,
                                       float* __restrict__ out,
                                       int R, int C) {
    int idx = blockIdx.x * blockDim.x + threadIdx.x;
    if (idx >= R * C) return;
    int r = idx / C;
    int c = idx % C;
    out[(size_t)c * R + r] = in[idx];
}

// ---------------------------------------------------------------------------
// Error-compensated f16 split: x = hi + lo (hi = f16(x), lo = f16(x - hi))
// ---------------------------------------------------------------------------
__global__ void MSGAD_split_f16_kernel(const float* __restrict__ x,
                                       _Float16* __restrict__ hi,
                                       _Float16* __restrict__ lo, int n) {
    int i = blockIdx.x * blockDim.x + threadIdx.x;
    if (i >= n) return;
    float v = x[i];
    _Float16 h = (_Float16)v;
    hi[i] = h;
    lo[i] = (_Float16)(v - (float)h);
}

// ---------------------------------------------------------------------------
// f32 WMMA GEMM (exact path, small GEMMs): Cmat = epi(A(MxK) @ Bt(NxK)^T)
// Each wave: 32x32 tile = 2x2 V_WMMA_F32_16X16X4_F32; 8 waves (2x4) / block.
// A frag (16x4 f32): lane L (half=L>>4, r=L&15): float2 at A[m+r][kk+2*half].
// B frag identical addressing on Bt. C/D elem v -> row m+v+8*half, col n+r.
// ---------------------------------------------------------------------------
__global__ __launch_bounds__(256)
void MSGAD_wmma_gemm_f32(const float* __restrict__ A,
                         const float* __restrict__ Bt,
                         const float* __restrict__ bias,
                         float* __restrict__ Cmat,
                         int K, int lda, int ldb, int ldc,
                         float alpha) {
    const int lane = threadIdx.x & 31;
    const int wave = threadIdx.x >> 5;
    const int half = lane >> 4;
    const int r    = lane & 15;

    const int m0 = blockIdx.y * 64  + (wave >> 2) * 32;
    const int n0 = blockIdx.x * 128 + (wave & 3) * 32;

    v8f c00 = {}; v8f c01 = {}; v8f c10 = {}; v8f c11 = {};

    const float* arow0 = A  + (size_t)(m0 + r)      * lda + 2 * half;
    const float* arow1 = A  + (size_t)(m0 + 16 + r) * lda + 2 * half;
    const float* brow0 = Bt + (size_t)(n0 + r)      * ldb + 2 * half;
    const float* brow1 = Bt + (size_t)(n0 + 16 + r) * ldb + 2 * half;

#pragma unroll 8
    for (int kk = 0; kk < K; kk += 4) {
        v2f a0 = *(const v2f*)(arow0 + kk);
        v2f a1 = *(const v2f*)(arow1 + kk);
        v2f b0 = *(const v2f*)(brow0 + kk);
        v2f b1 = *(const v2f*)(brow1 + kk);
        c00 = __builtin_amdgcn_wmma_f32_16x16x4_f32(false, a0, false, b0, (short)0, c00, false, false);
        c01 = __builtin_amdgcn_wmma_f32_16x16x4_f32(false, a0, false, b1, (short)0, c01, false, false);
        c10 = __builtin_amdgcn_wmma_f32_16x16x4_f32(false, a1, false, b0, (short)0, c10, false, false);
        c11 = __builtin_amdgcn_wmma_f32_16x16x4_f32(false, a1, false, b1, (short)0, c11, false, false);
    }

    const float bc0 = bias[n0 + r];
    const float bc1 = bias[n0 + 16 + r];
    const int row_lo = m0 + half * 8;
#pragma unroll
    for (int v = 0; v < 8; ++v) {
        Cmat[(size_t)(row_lo + v)      * ldc + (n0 + r)]      = fmaxf(c00[v] + bc0, 0.0f) * alpha;
        Cmat[(size_t)(row_lo + v)      * ldc + (n0 + 16 + r)] = fmaxf(c01[v] + bc1, 0.0f) * alpha;
        Cmat[(size_t)(row_lo + 16 + v) * ldc + (n0 + r)]      = fmaxf(c10[v] + bc0, 0.0f) * alpha;
        Cmat[(size_t)(row_lo + 16 + v) * ldc + (n0 + 16 + r)] = fmaxf(c11[v] + bc1, 0.0f) * alpha;
    }
}

// ---------------------------------------------------------------------------
// Split-f16 SYRK (dominant GEMM): C = A @ A^T with A = Ahi + Alo (f16 pair),
// C += Ahi·Bhi^T + Ahi·Blo^T + Alo·Bhi^T  via V_WMMA_F32_16X16X32_F16.
// 3 wmma per 32-K chunk vs 8 f32-wmma: 2.67x fewer matrix instructions.
// Each wave: 32x32 tile (2x2 accums); 8 waves (2x4) per 256-thr block.
//
// A frag (16-bit 16x32 ISA layout), lane L (half=L>>4, r=L&15):
//   v16h elems 0-7  = A[m+r][k0 + 8*half  .. +8)   (one b128)
//   v16h elems 8-15 = A[m+r][k0+16+8*half .. +8)   (one b128, +16 halfs)
// B frag (32x16), same K order as A, N=r:
//   elems 0-7 = Bt[n+r][k0+8*half ..+8), elems 8-15 = Bt[n+r][k0+16+8*half ..+8)
// ---------------------------------------------------------------------------
template <int K>
__global__ __launch_bounds__(256)
void MSGAD_wmma_syrk_f16x3(const _Float16* __restrict__ Ahi,
                           const _Float16* __restrict__ Alo,
                           float* __restrict__ Cmat, int ldc) {
    const int lane = threadIdx.x & 31;
    const int wave = threadIdx.x >> 5;
    const int half = lane >> 4;
    const int r    = lane & 15;

    const int m0 = blockIdx.y * 64  + (wave >> 2) * 32;
    const int n0 = blockIdx.x * 128 + (wave & 3) * 32;

    v8f acc[2][2];
#pragma unroll
    for (int i = 0; i < 2; ++i)
#pragma unroll
        for (int j = 0; j < 2; ++j) acc[i][j] = (v8f){};

    const _Float16* ah[2]; const _Float16* al[2];
    const _Float16* bh[2]; const _Float16* bl[2];
#pragma unroll
    for (int i = 0; i < 2; ++i) {
        size_t row = (size_t)(m0 + i * 16 + r) * K;
        ah[i] = Ahi + row + 8 * half;
        al[i] = Alo + row + 8 * half;
    }
#pragma unroll
    for (int j = 0; j < 2; ++j) {
        size_t row = (size_t)(n0 + j * 16 + r) * K;
        bh[j] = Ahi + row + 8 * half;
        bl[j] = Alo + row + 8 * half;
    }

#pragma unroll 2
    for (int k0 = 0; k0 < K; k0 += 32) {
        v16h fah[2], fal[2], fbh[2], fbl[2];
#pragma unroll
        for (int i = 0; i < 2; ++i) {
            v8h p0 = *(const v8h*)(ah[i] + k0);
            v8h p1 = *(const v8h*)(ah[i] + k0 + 16);
            fah[i] = __builtin_shufflevector(p0, p1, 0,1,2,3,4,5,6,7,8,9,10,11,12,13,14,15);
            v8h q0 = *(const v8h*)(al[i] + k0);
            v8h q1 = *(const v8h*)(al[i] + k0 + 16);
            fal[i] = __builtin_shufflevector(q0, q1, 0,1,2,3,4,5,6,7,8,9,10,11,12,13,14,15);
        }
#pragma unroll
        for (int j = 0; j < 2; ++j) {
            v8h p0 = *(const v8h*)(bh[j] + k0);
            v8h p1 = *(const v8h*)(bh[j] + k0 + 16);
            fbh[j] = __builtin_shufflevector(p0, p1, 0,1,2,3,4,5,6,7,8,9,10,11,12,13,14,15);
            v8h q0 = *(const v8h*)(bl[j] + k0);
            v8h q1 = *(const v8h*)(bl[j] + k0 + 16);
            fbl[j] = __builtin_shufflevector(q0, q1, 0,1,2,3,4,5,6,7,8,9,10,11,12,13,14,15);
        }
#pragma unroll
        for (int i = 0; i < 2; ++i)
#pragma unroll
            for (int j = 0; j < 2; ++j) {
                acc[i][j] = __builtin_amdgcn_wmma_f32_16x16x32_f16(false, fah[i], false, fbh[j], (short)0, acc[i][j], false, false);
                acc[i][j] = __builtin_amdgcn_wmma_f32_16x16x32_f16(false, fah[i], false, fbl[j], (short)0, acc[i][j], false, false);
                acc[i][j] = __builtin_amdgcn_wmma_f32_16x16x32_f16(false, fal[i], false, fbh[j], (short)0, acc[i][j], false, false);
            }
    }

    const int row_lo = m0 + half * 8;
#pragma unroll
    for (int i = 0; i < 2; ++i)
#pragma unroll
        for (int j = 0; j < 2; ++j)
#pragma unroll
            for (int v = 0; v < 8; ++v)
                Cmat[(size_t)(row_lo + i * 16 + v) * ldc + (n0 + j * 16 + r)] = acc[i][j][v];
}

// ---------------------------------------------------------------------------
// Pipeline:
//   _COEF == [5,0,0,0,0] (binomial identity) => all_h = 5*h exactly; the
//   whole graph-Laplacian part (deg/dinv/src/dst, 4 lap rounds) is a no-op.
//   h1     = relu(in_feat @ W1 + b1)   f32 WMMA  (8192x256x256)
//   all_h  = 5*relu(h1 @ W2 + b2)      f32 WMMA  (8192x256x128)
//   recons = all_h @ all_h^T           f16x3 WMMA (8192x8192x128, dominant)
//   d_out  = [recons (8192*8192) | all_h (8192*128)]
// ---------------------------------------------------------------------------
extern "C" void kernel_launch(void* const* d_in, const int* in_sizes, int n_in,
                              void* d_out, int out_size, void* d_ws, size_t ws_size,
                              hipStream_t stream) {
    const float* in_feat = (const float*)d_in[0];
    // d_in[1] = src, d_in[2] = dst : mathematically dead (COEF[1..4] == 0)
    const float* W1 = (const float*)d_in[3];
    const float* b1 = (const float*)d_in[4];
    const float* W2 = (const float*)d_in[5];
    const float* b2 = (const float*)d_in[6];

    const int Nn = 8192;   // nodes
    const int IN = 256;    // input features
    const int H2 = 256;    // 2*H
    const int H  = 128;    // hidden

    float* ws  = (float*)d_ws;
    float* W1t = ws;                          // H2 x IN  (65536 f)
    float* W2t = W1t + (size_t)IN * H2;       // H  x H2  (32768 f)
    float* h1  = W2t + (size_t)H2 * H;        // Nn x H2  (2097152 f)
    _Float16* Ahi = (_Float16*)(h1 + (size_t)Nn * H2);   // Nn x H halfs (2 MB)
    _Float16* Alo = Ahi + (size_t)Nn * H;                // Nn x H halfs (2 MB)

    float* recons = (float*)d_out;                     // Nn x Nn
    float* allh   = recons + (size_t)Nn * Nn;          // Nn x H

    // Transpose weights to N x K row-major so GEMMs read Bt with b64 loads
    MSGAD_transpose_kernel<<<(IN * H2 + 255) / 256, 256, 0, stream>>>(W1, W1t, IN, H2);
    MSGAD_transpose_kernel<<<(H2 * H + 255) / 256, 256, 0, stream>>>(W2, W2t, H2, H);

    // GEMM1: h1 = relu(in_feat @ W1 + b1)   M=8192 N=256 K=256
    {
        dim3 grid(H2 / 128, Nn / 64);
        MSGAD_wmma_gemm_f32<<<grid, 256, 0, stream>>>(
            in_feat, W1t, b1, h1, IN, IN, IN, H2, 1.0f);
    }
    // GEMM2: allh = 5 * relu(h1 @ W2 + b2)  M=8192 N=128 K=256
    {
        dim3 grid(H / 128, Nn / 64);
        MSGAD_wmma_gemm_f32<<<grid, 256, 0, stream>>>(
            h1, W2t, b2, allh, H2, H2, H2, H, 5.0f);
    }
    // Split allh into f16 hi/lo for the compensated SYRK
    {
        int n = Nn * H;
        MSGAD_split_f16_kernel<<<(n + 255) / 256, 256, 0, stream>>>(allh, Ahi, Alo, n);
    }
    // GEMM3: recons = allh @ allh^T   M=N=8192 K=128, f16x3 compensated WMMA
    {
        dim3 grid(Nn / 128, Nn / 64);
        MSGAD_wmma_syrk_f16x3<128><<<grid, 256, 0, stream>>>(Ahi, Alo, recons, Nn);
    }
}
